// GraphUNetModel_66228395705233
// MI455X (gfx1250) — matrix operand order repaired and checked by reading, not compiled, not run on hardware
//
#include <hip/hip_runtime.h>
#include <hip/hip_bf16.h>

#define DEV __device__ __forceinline__

typedef __attribute__((ext_vector_type(2)))  float  v2f;
typedef __attribute__((ext_vector_type(8)))  float  v8f;
typedef __attribute__((ext_vector_type(16))) __bf16 v16bf;
typedef int v4i __attribute__((vector_size(16)));
typedef __attribute__((address_space(1))) v4i* as1_v4i_ptr;
typedef __attribute__((address_space(3))) v4i* as3_v4i_ptr;

// Model dims (fixed by the reference)
constexpr int B_    = 256;
constexpr int N_    = 256;
constexpr int FATOM = 9;
constexpr int VOC   = 128;
constexpr int EMB_  = 256;
constexpr int HID_  = 32;

// ---------------------------------------------------------------------------
// WMMA helpers
// ---------------------------------------------------------------------------
DEV v8f wmma4_f32(v2f a, v2f b, v8f c, int lo, int hi) {
#if __has_builtin(__builtin_amdgcn_wmma_f32_16x16x4_f32)
    (void)lo; (void)hi;
    return __builtin_amdgcn_wmma_f32_16x16x4_f32(false, a, false, b,
                                                 (short)0, c, false, false);
#else
    float ar[2] = {a[0], a[1]}, br[2] = {b[0], b[1]};
    #pragma unroll
    for (int e = 0; e < 8; ++e) {
        int M = e + 8 * hi;
        float acc = c[e];
        #pragma unroll
        for (int k = 0; k < 4; ++k) {
            float av = __shfl(ar[k & 1], M  + 16 * (k >> 1), 32);
            float bv = __shfl(br[k & 1], lo + 16 * (k >> 1), 32);
            acc = fmaf(av, bv, acc);
        }
        c[e] = acc;
    }
    return c;
#endif
}

// ---------------------------------------------------------------------------
// Async global->LDS copy (gfx1250 GLOBAL_LOAD_ASYNC_TO_LDS_B128, ASYNCcnt).
// Param types per hipcc diagnostic: (int4 AS1*, int4 AS3*, imm, imm).
// ---------------------------------------------------------------------------
DEV void cp_b128(const float* g, float* l) {
#if __has_builtin(__builtin_amdgcn_global_load_async_to_lds_b128)
    __builtin_amdgcn_global_load_async_to_lds_b128(
        (as1_v4i_ptr)g, (as3_v4i_ptr)l, 0, 0);
#else
    *(float4*)l = *(const float4*)g;
#endif
}

DEV void async_wait() {
#if __has_builtin(__builtin_amdgcn_global_load_async_to_lds_b128)
  #if __has_builtin(__builtin_amdgcn_s_wait_asynccnt)
    __builtin_amdgcn_s_wait_asynccnt(0);
  #else
    asm volatile("s_wait_asynccnt 0" ::: "memory");
  #endif
#endif
}

// ---------------------------------------------------------------------------
// 1) AtomEncoder: h[b,n,e] = sum_f table[f, x_atom[b,n,f], e]
// ---------------------------------------------------------------------------
__global__ void k_atom_encoder(const int* __restrict__ xa,
                               const float* __restrict__ table,
                               float* __restrict__ h) {
    int bn = blockIdx.x;
    int e  = threadIdx.x;
    const int* row = xa + (size_t)bn * FATOM;
    float acc = 0.f;
    #pragma unroll
    for (int f = 0; f < FATOM; ++f) {
        int id = row[f];
        acc += table[((size_t)f * VOC + id) * EMB_ + e];
    }
    h[(size_t)bn * EMB_ + e] = acc;
}

// ---------------------------------------------------------------------------
// 2) di[b,i] = rsqrt(2 + rowsum(A))
// ---------------------------------------------------------------------------
__global__ void k_deg_rsqrt(const float* __restrict__ A,
                            float* __restrict__ di, int n) {
    int t = blockIdx.x * blockDim.x + threadIdx.x;
    if (t >= B_ * n) return;
    int b = t / n, i = t % n;
    const float* row = A + ((size_t)b * n + i) * n;
    float d = 2.0f;
    for (int j = 0; j < n; ++j) d += row[j];
    di[t] = rsqrtf(d);
}

// ---------------------------------------------------------------------------
// 3) Z = X @ W  (X:[B,n,K], W:[K,m]) — f32 WMMA 16x16x4, 1 wave / 16x16 tile
// ---------------------------------------------------------------------------
__global__ void k_matmul_wmma(const float* __restrict__ X,
                              const float* __restrict__ W,
                              float* __restrict__ Z, int n, int K, int m) {
    int b    = blockIdx.y;
    int mt   = m >> 4;
    int rt   = blockIdx.x / mt, ct = blockIdx.x % mt;
    int lane = threadIdx.x, lo = lane & 15, hi = lane >> 4;
    const float* Xb = X + (size_t)b * n * K;
    int row = rt * 16 + lo;
    int col = ct * 16 + lo;
    v8f acc = {};
    for (int k0 = 0; k0 < K; k0 += 4) {
        int ka = k0 + 2 * hi;
        v2f a, bb;
        a[0]  = Xb[(size_t)row * K + ka];
        a[1]  = Xb[(size_t)row * K + ka + 1];
        bb[0] = W[(size_t)ka * m + col];
        bb[1] = W[(size_t)(ka + 1) * m + col];
        acc = wmma4_f32(a, bb, acc, lo, hi);
    }
    float* Zb = Z + (size_t)b * n * m;
    #pragma unroll
    for (int e = 0; e < 8; ++e) {
        int r = rt * 16 + e + 8 * hi;
        Zb[(size_t)r * m + col] = acc[e];
    }
}

// ---------------------------------------------------------------------------
// 4) GCN propagate: Y = di .* ((A + 2I)(di .* Z)) + bias, optional relu
// ---------------------------------------------------------------------------
__global__ void k_gcn_spmm_wmma(const float* __restrict__ A,
                                const float* __restrict__ di,
                                const float* __restrict__ Z,
                                const float* __restrict__ bias,
                                float* __restrict__ Y, int n, int relu) {
    const int m = HID_;
    int b    = blockIdx.y;
    int mt   = m >> 4;
    int rt   = blockIdx.x / mt, ct = blockIdx.x % mt;
    int lane = threadIdx.x, lo = lane & 15, hi = lane >> 4;
    const float* Ab  = A  + (size_t)b * n * n;
    const float* Zb  = Z  + (size_t)b * n * m;
    const float* dib = di + (size_t)b * n;
    int row = rt * 16 + lo;
    int col = ct * 16 + lo;
    v8f acc = {};
    for (int k0 = 0; k0 < n; k0 += 4) {
        int ka = k0 + 2 * hi;
        float a0 = Ab[(size_t)row * n + ka];
        float a1 = Ab[(size_t)row * n + ka + 1];
        if (row == ka)     a0 += 2.0f;
        if (row == ka + 1) a1 += 2.0f;
        a0 *= dib[ka];
        a1 *= dib[ka + 1];
        v2f a, bb;
        a[0] = a0; a[1] = a1;
        bb[0] = Zb[(size_t)ka * m + col];
        bb[1] = Zb[(size_t)(ka + 1) * m + col];
        acc = wmma4_f32(a, bb, acc, lo, hi);
    }
    float bv = bias[col];
    float* Yb = Y + (size_t)b * n * m;
    #pragma unroll
    for (int e = 0; e < 8; ++e) {
        int r = rt * 16 + e + 8 * hi;
        float v = acc[e] * dib[r] + bv;
        if (relu) v = fmaxf(v, 0.0f);
        Yb[(size_t)r * m + col] = v;
    }
}

// ---------------------------------------------------------------------------
// 5) augment: Out = ((A+I)@(A+I)) with zeroed diagonal — bf16 WMMA 16x16x32.
//    8 waves / block, 64x64 macro-tile, LDS-staged panels via async-to-LDS.
// ---------------------------------------------------------------------------
constexpr int ALD = 36;   // 64x32 A panel, padded row (16B-aligned, bank-safe)
constexpr int BLD = 68;   // 32x64 B panel, padded row

__global__ void k_augment_wmma(const float* __restrict__ A,
                               float* __restrict__ Out, int n) {
    __shared__ __align__(16) float As[64 * ALD];
    __shared__ __align__(16) float Bs[32 * BLD];
    int b  = blockIdx.y;
    int nt = n >> 6;
    int rt = blockIdx.x / nt, ct = blockIdx.x % nt;
    int rowBase = rt * 64, colBase = ct * 64;
    int t    = threadIdx.x;
    int lane = t & 31, w = t >> 5;
    int lo = lane & 15, hi = (lane >> 4) & 1;
    int band  = (w >> 1) * 16;          // row band inside macro-tile (4 bands)
    int chalf = (w & 1) * 32;           // this wave's two col tiles
    const float* Ab = A + (size_t)b * n * n;

    v8f acc0 = {}, acc1 = {};

    for (int k0 = 0; k0 < n; k0 += 32) {
        // ---- stage A panel 64x32 (512 float4) and B panel 32x64 (512 float4)
        #pragma unroll
        for (int q = t; q < 512; q += 256) {
            int r = q >> 3, s = q & 7;
            cp_b128(Ab + (size_t)(rowBase + r) * n + k0 + 4 * s,
                    &As[r * ALD + 4 * s]);
        }
        #pragma unroll
        for (int q = t; q < 512; q += 256) {
            int r = q >> 4, s = q & 15;
            cp_b128(Ab + (size_t)(k0 + r) * n + colBase + 4 * s,
                    &Bs[r * BLD + 4 * s]);
        }
        if (k0 + 32 < n) {   // prefetch next K panels into cache hierarchy
            __builtin_prefetch(Ab + (size_t)(rowBase + (t >> 3)) * n + (k0 + 32) + 4 * (t & 7), 0, 3);
            __builtin_prefetch(Ab + (size_t)(k0 + 32 + (t >> 4)) * n + colBase + 4 * (t & 15), 0, 3);
        }
        async_wait();
        __syncthreads();

        // ---- build bf16 fragments from LDS (identity fused on the fly) ----
        v16bf av, bv0, bv1;
        #pragma unroll
        for (int e = 0; e < 16; ++e) {
            int ka = ((e < 8) ? 0 : 16) + hi * 8 + (e & 7);     // A frag K index
            int gr = rowBase + band + lo;
            float va = As[(band + lo) * ALD + ka] + ((gr == k0 + ka) ? 1.0f : 0.0f);
            av[e] = (__bf16)va;
            int kb  = ((e < 8) ? 0 : 16) + hi * 8 + (e & 7);    // B frag K index
            int gk  = k0 + kb;
            int gc0 = colBase + chalf + lo;
            float v0 = Bs[kb * BLD + chalf + lo]      + ((gk == gc0)      ? 1.0f : 0.0f);
            float v1 = Bs[kb * BLD + chalf + 16 + lo] + ((gk == gc0 + 16) ? 1.0f : 0.0f);
            bv0[e] = (__bf16)v0;
            bv1[e] = (__bf16)v1;
        }
        acc0 = __builtin_amdgcn_wmma_f32_16x16x32_bf16(false, av, false, bv0,
                                                       (short)0, acc0, false, false);
        acc1 = __builtin_amdgcn_wmma_f32_16x16x32_bf16(false, av, false, bv1,
                                                       (short)0, acc1, false, false);
        __syncthreads();
    }

    float* Ob = Out + (size_t)b * n * n;
    #pragma unroll
    for (int e = 0; e < 8; ++e) {
        int r  = rowBase + band + e + 8 * hi;
        int c0 = colBase + chalf + lo;
        int c1 = c0 + 16;
        Ob[(size_t)r * n + c0] = (r == c0) ? 0.0f : acc0[e];
        Ob[(size_t)r * n + c1] = (r == c1) ? 0.0f : acc1[e];
    }
}

// ---------------------------------------------------------------------------
// 6) TopK pooling: score = tanh(x.w/||w||); stable descending rank selection.
// ---------------------------------------------------------------------------
__global__ void k_topk(const float* __restrict__ X, const float* __restrict__ w,
                       float* __restrict__ Xg, int* __restrict__ idx,
                       int n, int k) {
    __shared__ float sc[256];
    int b = blockIdx.x, i = threadIdx.x;
    float wn = 0.f;
    #pragma unroll
    for (int c = 0; c < HID_; ++c) wn += w[c] * w[c];
    wn = sqrtf(wn);
    float si = 0.f;
    if (i < n) {
        const float* xr = X + ((size_t)b * n + i) * HID_;
        float s = 0.f;
        #pragma unroll
        for (int c = 0; c < HID_; ++c) s += xr[c] * w[c];
        si = tanhf(s / wn);
        sc[i] = si;
    }
    __syncthreads();
    if (i >= n) return;
    int r = 0;
    for (int j = 0; j < n; ++j) {
        float sj = sc[j];
        if (sj > si || (sj == si && j < i)) r++;
    }
    if (r < k) {
        idx[(size_t)b * k + r] = i;
        const float* xr = X + ((size_t)b * n + i) * HID_;
        float* xo = Xg + ((size_t)b * k + r) * HID_;
        #pragma unroll
        for (int c = 0; c < HID_; ++c) xo[c] = xr[c] * si;
    }
}

// ---------------------------------------------------------------------------
// 7) Gather pooled adjacency: Ad[b,p,q] = As[b, idx[p], idx[q]]
// ---------------------------------------------------------------------------
__global__ void k_gatherA(const float* __restrict__ As, const int* __restrict__ idx,
                          float* __restrict__ Ad, int n, int k) {
    size_t t = (size_t)blockIdx.x * blockDim.x + threadIdx.x;
    size_t total = (size_t)B_ * k * k;
    if (t >= total) return;
    int q = (int)(t % k);
    size_t tt = t / k;
    int p = (int)(tt % k);
    int b = (int)(tt / k);
    Ad[t] = As[((size_t)b * n + idx[(size_t)b * k + p]) * n + idx[(size_t)b * k + q]];
}

__global__ void k_copy(const float* __restrict__ src, float* __restrict__ dst,
                       size_t count) {
    size_t t = (size_t)blockIdx.x * blockDim.x + threadIdx.x;
    if (t < count) dst[t] = src[t];
}

__global__ void k_scatter_add(float* __restrict__ X, const float* __restrict__ Xs,
                              const int* __restrict__ idx, int n, int k) {
    size_t t = (size_t)blockIdx.x * blockDim.x + threadIdx.x;
    size_t total = (size_t)B_ * k * HID_;
    if (t >= total) return;
    int c = (int)(t % HID_);
    size_t rr = t / HID_;
    int p = (int)(rr % k);
    int b = (int)(rr / k);
    X[((size_t)b * n + idx[(size_t)b * k + p]) * HID_ + c] += Xs[t];
}

// ---------------------------------------------------------------------------
// Collapsed final stage: sum_i(An @ (x@W) + b)  ==  ((g^T x) W) + N*b
// ---------------------------------------------------------------------------
__global__ void k_gvec(const float* __restrict__ adj, const float* __restrict__ di,
                       float* __restrict__ g) {
    int t = blockIdx.x * blockDim.x + threadIdx.x;
    if (t >= B_ * N_) return;
    int b = t / N_, kk = t % N_;
    const float* row = adj + ((size_t)b * N_ + kk) * N_;
    const float* dib = di + (size_t)b * N_;
    float s = 2.0f * dib[kk];
    for (int i = 0; i < N_; ++i) s += row[i] * dib[i];
    g[t] = dib[kk] * s;
}

__global__ void k_tvec(const float* __restrict__ g, const float* __restrict__ X,
                       float* __restrict__ t) {
    int tid = blockIdx.x * blockDim.x + threadIdx.x;
    if (tid >= B_ * HID_) return;
    int b = tid / HID_, c = tid % HID_;
    const float* gb = g + (size_t)b * N_;
    const float* Xb = X + (size_t)b * N_ * HID_;
    float s = 0.f;
    for (int kk = 0; kk < N_; ++kk) s += gb[kk] * Xb[(size_t)kk * HID_ + c];
    t[tid] = s;
}

__global__ void k_out(const float* __restrict__ t, const float* __restrict__ Wl,
                      const float* __restrict__ bl, float* __restrict__ out) {
    int tid = blockIdx.x * blockDim.x + threadIdx.x;
    if (tid >= B_ * EMB_) return;
    int b = tid / EMB_, e = tid % EMB_;
    const float* tb = t + (size_t)b * HID_;
    float s = 256.0f * bl[e];
    #pragma unroll
    for (int c = 0; c < HID_; ++c) s += tb[c] * Wl[(size_t)c * EMB_ + e];
    out[tid] = s;
}

// ---------------------------------------------------------------------------
// Orchestration
// ---------------------------------------------------------------------------
extern "C" void kernel_launch(void* const* d_in, const int* in_sizes, int n_in,
                              void* d_out, int out_size, void* d_ws, size_t ws_size,
                              hipStream_t stream) {
    const float* adj      = (const float*)d_in[0];
    const int*   x_atom   = (const int*)  d_in[1];
    const float* table    = (const float*)d_in[2];
    const float* w_down0  = (const float*)d_in[3];
    const float* b_down0  = (const float*)d_in[4];
    const float* w_down   = (const float*)d_in[5];
    const float* b_down   = (const float*)d_in[6];
    const float* w_pool   = (const float*)d_in[7];
    const float* w_up     = (const float*)d_in[8];
    const float* b_up     = (const float*)d_in[9];
    const float* w_last   = (const float*)d_in[10];
    const float* b_last   = (const float*)d_in[11];
    float* out = (float*)d_out;

    float* wsf = (float*)d_ws;
    constexpr size_t SZ_BIG = (size_t)B_ * N_ * N_;
    size_t off = 0;
    float* bigA = wsf + off; off += SZ_BIG;
    float* x0   = wsf + off; off += (size_t)B_ * 256 * HID_;
    float* x1   = wsf + off; off += (size_t)B_ * 128 * HID_;
    float* x2   = wsf + off; off += (size_t)B_ *  64 * HID_;
    float* xa   = wsf + off; off += (size_t)B_ * 256 * HID_;
    float* xb   = wsf + off; off += (size_t)B_ * 256 * HID_;
    float* zt   = wsf + off; off += (size_t)B_ * 256 * HID_;
    float* A1p  = wsf + off; off += (size_t)B_ * 128 * 128;
    float* A2p  = wsf + off; off += (size_t)B_ *  64 *  64;
    float* A3p  = wsf + off; off += (size_t)B_ *  32 *  32;
    float* di0  = wsf + off; off += (size_t)B_ * 256;
    float* di1  = wsf + off; off += (size_t)B_ * 128;
    float* di2  = wsf + off; off += (size_t)B_ *  64;
    float* di3  = wsf + off; off += (size_t)B_ *  32;
    float* gv   = wsf + off; off += (size_t)B_ * 256;
    float* tv   = wsf + off; off += (size_t)B_ * HID_;
    int* idx0 = (int*)(wsf + off);
    int* idx1 = idx0 + (size_t)B_ * 128;
    int* idx2 = idx1 + (size_t)B_ * 64;

    auto cdiv = [](size_t a, size_t b) { return (unsigned)((a + b - 1) / b); };

    // Encoder + level-0 GCN
    k_atom_encoder<<<B_ * N_, EMB_, 0, stream>>>(x_atom, table, bigA);
    k_deg_rsqrt<<<cdiv((size_t)B_ * 256, 256), 256, 0, stream>>>(adj, di0, 256);
    k_matmul_wmma<<<dim3((256 / 16) * (HID_ / 16), B_), 32, 0, stream>>>(
        bigA, w_down0, zt, 256, EMB_, HID_);
    k_gcn_spmm_wmma<<<dim3((256 / 16) * 2, B_), 32, 0, stream>>>(
        adj, di0, zt, b_down0, x0, 256, 1);

    // down level 0: n=256 -> 128
    k_augment_wmma<<<dim3((256 / 64) * (256 / 64), B_), 256, 0, stream>>>(adj, bigA, 256);
    k_topk<<<B_, 256, 0, stream>>>(x0, w_pool + 0 * HID_, xa, idx0, 256, 128);
    k_gatherA<<<cdiv((size_t)B_ * 128 * 128, 256), 256, 0, stream>>>(bigA, idx0, A1p, 256, 128);
    k_deg_rsqrt<<<cdiv((size_t)B_ * 128, 256), 256, 0, stream>>>(A1p, di1, 128);
    k_matmul_wmma<<<dim3((128 / 16) * 2, B_), 32, 0, stream>>>(xa, w_down + 0 * HID_ * HID_, zt, 128, HID_, HID_);
    k_gcn_spmm_wmma<<<dim3((128 / 16) * 2, B_), 32, 0, stream>>>(A1p, di1, zt, b_down + 0 * HID_, x1, 128, 1);

    // down level 1: n=128 -> 64
    k_augment_wmma<<<dim3((128 / 64) * (128 / 64), B_), 256, 0, stream>>>(A1p, bigA, 128);
    k_topk<<<B_, 256, 0, stream>>>(x1, w_pool + 1 * HID_, xa, idx1, 128, 64);
    k_gatherA<<<cdiv((size_t)B_ * 64 * 64, 256), 256, 0, stream>>>(bigA, idx1, A2p, 128, 64);
    k_deg_rsqrt<<<cdiv((size_t)B_ * 64, 256), 256, 0, stream>>>(A2p, di2, 64);
    k_matmul_wmma<<<dim3((64 / 16) * 2, B_), 32, 0, stream>>>(xa, w_down + 1 * HID_ * HID_, zt, 64, HID_, HID_);
    k_gcn_spmm_wmma<<<dim3((64 / 16) * 2, B_), 32, 0, stream>>>(A2p, di2, zt, b_down + 1 * HID_, x2, 64, 1);

    // down level 2: n=64 -> 32
    k_augment_wmma<<<dim3(1, B_), 256, 0, stream>>>(A2p, bigA, 64);
    k_topk<<<B_, 256, 0, stream>>>(x2, w_pool + 2 * HID_, xa, idx2, 64, 32);
    k_gatherA<<<cdiv((size_t)B_ * 32 * 32, 256), 256, 0, stream>>>(bigA, idx2, A3p, 64, 32);
    k_deg_rsqrt<<<cdiv((size_t)B_ * 32, 256), 256, 0, stream>>>(A3p, di3, 32);
    k_matmul_wmma<<<dim3((32 / 16) * 2, B_), 32, 0, stream>>>(xa, w_down + 2 * HID_ * HID_, zt, 32, HID_, HID_);
    k_gcn_spmm_wmma<<<dim3((32 / 16) * 2, B_), 32, 0, stream>>>(A3p, di3, zt, b_down + 2 * HID_, xb, 32, 1);

    // up i=0 (j=2)
    k_copy<<<cdiv((size_t)B_ * 64 * HID_, 256), 256, 0, stream>>>(x2, xa, (size_t)B_ * 64 * HID_);
    k_scatter_add<<<cdiv((size_t)B_ * 32 * HID_, 256), 256, 0, stream>>>(xa, xb, idx2, 64, 32);
    k_matmul_wmma<<<dim3((64 / 16) * 2, B_), 32, 0, stream>>>(xa, w_up + 0 * HID_ * HID_, zt, 64, HID_, HID_);
    k_gcn_spmm_wmma<<<dim3((64 / 16) * 2, B_), 32, 0, stream>>>(A2p, di2, zt, b_up + 0 * HID_, xb, 64, 1);

    // up i=1 (j=1)
    k_copy<<<cdiv((size_t)B_ * 128 * HID_, 256), 256, 0, stream>>>(x1, xa, (size_t)B_ * 128 * HID_);
    k_scatter_add<<<cdiv((size_t)B_ * 64 * HID_, 256), 256, 0, stream>>>(xa, xb, idx1, 128, 64);
    k_matmul_wmma<<<dim3((128 / 16) * 2, B_), 32, 0, stream>>>(xa, w_up + 1 * HID_ * HID_, zt, 128, HID_, HID_);
    k_gcn_spmm_wmma<<<dim3((128 / 16) * 2, B_), 32, 0, stream>>>(A1p, di1, zt, b_up + 1 * HID_, xb, 128, 1);

    // up i=2 (j=0) + collapsed final stage
    k_copy<<<cdiv((size_t)B_ * 256 * HID_, 256), 256, 0, stream>>>(x0, xa, (size_t)B_ * 256 * HID_);
    k_scatter_add<<<cdiv((size_t)B_ * 128 * HID_, 256), 256, 0, stream>>>(xa, xb, idx0, 256, 128);
    k_gvec<<<cdiv((size_t)B_ * N_, 256), 256, 0, stream>>>(adj, di0, gv);
    k_tvec<<<cdiv((size_t)B_ * HID_, 256), 256, 0, stream>>>(gv, xa, tv);
    k_out<<<cdiv((size_t)B_ * EMB_, 256), 256, 0, stream>>>(tv, w_last, b_last, out);
}
